// PiInterferenceLayer_35880156790976
// MI455X (gfx1250) — matrix operand order repaired and checked by reading, not compiled, not run on hardware
//
#include <hip/hip_runtime.h>
#include <hip/hip_bf16.h>

#define DIM   1024
#define NPAT  4096
#define BATCH 8192

typedef __attribute__((ext_vector_type(16))) _Float16 v16h;
typedef __attribute__((ext_vector_type(8)))  _Float16 v8h;
typedef __attribute__((ext_vector_type(8)))  float    v8f;

// ---------------- prep: convert x (f32 -> f16) ----------------
__global__ void cvt_x_kernel(const float* __restrict__ xr, const float* __restrict__ xi,
                             _Float16* __restrict__ Xr, _Float16* __restrict__ Xi, int n) {
    int i = blockIdx.x * blockDim.x + threadIdx.x;
    if (i < n) {
        Xr[i] = (_Float16)xr[i];
        Xi[i] = (_Float16)xi[i];
    }
}

// ---------------- prep: phase-shift patterns -> f16 ----------------
// Sr = pr*cos(phi) - pi*sin(phi) ; Si = pr*sin(phi) + pi*cos(phi)
__global__ void shift_kernel(const float* __restrict__ pr, const float* __restrict__ pim,
                             const float* __restrict__ phases,
                             _Float16* __restrict__ Sr, _Float16* __restrict__ Si, int n) {
    int i = blockIdx.x * blockDim.x + threadIdx.x;
    if (i < n) {
        int p = i >> 10;              // / DIM
        float s, c;
        __sincosf(phases[p], &s, &c);
        float a = pr[i], b = pim[i];
        Sr[i] = (_Float16)(a * c - b * s);
        Si[i] = (_Float16)(a * s + b * c);
    }
}

// ---------------- fragment loaders (per CDNA5 ISA VGPR layouts) ----------------
// 16-bit A fragment 16x32: elems 0..7 <- k = kb + (lane/16)*8 + 0..7
//                          elems 8..15 <- k = kb + 16 + (lane/16)*8 + 0..7
__device__ __forceinline__ v16h load_a_frag(const _Float16* p) {
    v8h lo = *(const v8h*)(p);
    v8h hi = *(const v8h*)(p + 16);
    return __builtin_shufflevector(lo, hi, 0,1,2,3,4,5,6,7,8,9,10,11,12,13,14,15);
}
// 16-bit B fragment 32x16: elems 0..7 <- k = kb + (lane/16)*8 + 0..7, elems 8..15 <- k = kb + 16 + (lane/16)*8 + 0..7
__device__ __forceinline__ v16h load_b_frag(const _Float16* p) {
    v8h lo = *(const v8h*)(p);
    v8h hi = *(const v8h*)(p + 16);
    return __builtin_shufflevector(lo, hi, 0,1,2,3,4,5,6,7,8,9,10,11,12,13,14,15);
}

__device__ __forceinline__ v8f wmma_f16(v16h a, v16h b, v8f c) {
    return __builtin_amdgcn_wmma_f32_16x16x32_f16(
        /*neg_a=*/false, a, /*neg_b=*/false, b,
        /*c_mod=*/(short)0, c, /*reuse_a=*/false, /*reuse_b=*/false);
}

// ---------------- main GEMM: out[b,p] = zr^2 + zi^2 ----------------
// Per wave: 32(M) x 32(N) register tile (2x2 WMMA tiles).
// Block = 8 waves arranged 2(M) x 4(N) -> 64 x 128 block tile.
__global__ __launch_bounds__(256)
void PiInterference_wmma_kernel(const _Float16* __restrict__ Xr, const _Float16* __restrict__ Xi,
                                const _Float16* __restrict__ Sr, const _Float16* __restrict__ Si,
                                float* __restrict__ out) {
    const int lane = threadIdx.x & 31;
    const int wid  = threadIdx.x >> 5;
    const int wm   = wid & 1;         // 2 waves along M
    const int wn   = wid >> 1;        // 4 waves along N
    const int m0   = blockIdx.x * 64  + wm * 32;
    const int n0   = blockIdx.y * 128 + wn * 32;
    const int lr   = lane & 15;       // row/col within 16-wide tile
    const int lh   = lane >> 4;       // half-wave select

    // A row pointers: [mtile][0]=Xr, [mtile][1]=Xi
    const _Float16* ap[2][2];
    // B row pointers: [ntile][0]=Sr, [ntile][1]=Si
    const _Float16* bp[2][2];
    #pragma unroll
    for (int t = 0; t < 2; ++t) {
        ap[t][0] = Xr + (size_t)(m0 + t * 16 + lr) * DIM + lh * 8;
        ap[t][1] = Xi + (size_t)(m0 + t * 16 + lr) * DIM + lh * 8;
        bp[t][0] = Sr + (size_t)(n0 + t * 16 + lr) * DIM + lh * 8;
        bp[t][1] = Si + (size_t)(n0 + t * 16 + lr) * DIM + lh * 8;
    }

    v8f accr[2][2] = {};   // zr         accumulators [mtile][ntile]
    v8f accp[2][2] = {};   // Xr*Si      accumulators
    v8f accn[2][2] = {};   // Xi*Sr      accumulators (zi = accp - accn)

    for (int k = 0; k < DIM; k += 32) {
        v16h a[2][2], b[2][2];
        #pragma unroll
        for (int t = 0; t < 2; ++t) {
            a[t][0] = load_a_frag(ap[t][0] + k);
            a[t][1] = load_a_frag(ap[t][1] + k);
            b[t][0] = load_b_frag(bp[t][0] + k);
            b[t][1] = load_b_frag(bp[t][1] + k);
        }
        #pragma unroll
        for (int mt = 0; mt < 2; ++mt) {
            #pragma unroll
            for (int nt = 0; nt < 2; ++nt) {
                accr[mt][nt] = wmma_f16(a[mt][0], b[nt][0], accr[mt][nt]);
                accr[mt][nt] = wmma_f16(a[mt][1], b[nt][1], accr[mt][nt]);
                accp[mt][nt] = wmma_f16(a[mt][0], b[nt][1], accp[mt][nt]);
                accn[mt][nt] = wmma_f16(a[mt][1], b[nt][0], accn[mt][nt]);
            }
        }
    }

    // C/D layout: VGPR r holds M = r + lh*8, N = lr
    #pragma unroll
    for (int mt = 0; mt < 2; ++mt) {
        #pragma unroll
        for (int r = 0; r < 8; ++r) {
            const size_t row = (size_t)(m0 + mt * 16 + lh * 8 + r) * NPAT;
            #pragma unroll
            for (int nt = 0; nt < 2; ++nt) {
                float zr = accr[mt][nt][r];
                float zi = accp[mt][nt][r] - accn[mt][nt][r];
                out[row + n0 + nt * 16 + lr] = zr * zr + zi * zi;
            }
        }
    }
}

extern "C" void kernel_launch(void* const* d_in, const int* in_sizes, int n_in,
                              void* d_out, int out_size, void* d_ws, size_t ws_size,
                              hipStream_t stream) {
    const float* xr = (const float*)d_in[0];
    const float* xi = (const float*)d_in[1];
    const float* pr = (const float*)d_in[2];
    const float* pi = (const float*)d_in[3];
    const float* ph = (const float*)d_in[4];
    float* out = (float*)d_out;

    // workspace layout: Xr, Xi (f16 BATCH*DIM each), Sr, Si (f16 NPAT*DIM each) ~ 50.3 MB
    _Float16* Xr = (_Float16*)d_ws;
    _Float16* Xi = Xr + (size_t)BATCH * DIM;
    _Float16* Sr = Xi + (size_t)BATCH * DIM;
    _Float16* Si = Sr + (size_t)NPAT * DIM;

    const int nx = BATCH * DIM;
    cvt_x_kernel<<<(nx + 255) / 256, 256, 0, stream>>>(xr, xi, Xr, Xi, nx);

    const int np = NPAT * DIM;
    shift_kernel<<<(np + 255) / 256, 256, 0, stream>>>(pr, pi, ph, Sr, Si, np);

    dim3 grid(BATCH / 64, NPAT / 128);   // 128 x 32 blocks
    PiInterference_wmma_kernel<<<grid, 256, 0, stream>>>(Xr, Xi, Sr, Si, out);
}
